// WindowAttention_8555574854490
// MI455X (gfx1250) — matrix operand order repaired and checked by reading, not compiled, not run on hardware
//
#include <hip/hip_runtime.h>
#include <hip/hip_bf16.h>
#include <stdint.h>

typedef __attribute__((ext_vector_type(16))) _Float16 v16h;
typedef __attribute__((ext_vector_type(8)))  float    v8f;

#define WA_B    8192
#define WA_NW   1024
#define WA_H    12
#define WA_N    49
#define WA_C    384
#define WA_D    32
#define WA_ROWS (WA_B * WA_N)          // 401408
#define WA_SCALE 0.17677669529663687f  // 32^-0.5

// LDS row strides (halfs): multiples of 8 (16B-aligned chunks) whose dword stride
// is coprime-ish with 64 banks (20 dwords, 36 dwords -> 16 distinct row starts).
#define AP 40
#define TP 72

__device__ __forceinline__ uint32_t lds_of(const void* p) {
  return (uint32_t)(uintptr_t)p;   // low 32 bits of generic LDS pointer == LDS offset
}

// ---------- WMMA fragment helpers (layouts per CDNA5 ISA) ----------

// 16-bit A-matrix 16x32: lanes 0-15 row M=lane, halves h0-7 -> K0-7, h8-15 -> K16-23
//                        lanes 16-31: h0-7 -> K8-15, h8-15 -> K24-31
__device__ __forceinline__ v16h frag_a(const _Float16* rowp, int lane) {
  v16h a;
  const int lo = (lane & 16) ? 8 : 0;
#pragma unroll
  for (int h = 0; h < 8; ++h) {
    a[h]     = rowp[lo + h];
    a[h + 8] = rowp[lo + 16 + h];
  }
  return a;
}

// 16-bit B-matrix 32x16: lane = column n (mod 16); lanes 0-15: h0-7 -> K0-7, h8-15 -> K16-23; lanes 16-31: h0-7 -> K8-15, h8-15 -> K24-31.
// rowp = B-source row n, where B-source[n][k] == B[k][n].
__device__ __forceinline__ v16h frag_b(const _Float16* rowp, int lane) {
  v16h b;
  const int ko = (lane & 16) ? 8 : 0;
#pragma unroll
  for (int h = 0; h < 8; ++h) {
    b[h]     = rowp[ko + h];
    b[h + 8] = rowp[ko + 16 + h];
  }
  return b;
}

__device__ __forceinline__ v8f wmma_f16(v16h a, v16h b, v8f c) {
  return __builtin_amdgcn_wmma_f32_16x16x32_f16(false, a, false, b, (short)0, c,
                                                false, false);
}

// ---------- CDNA5 async global->LDS copies (CDNA5 ISA) ----------
__device__ __forceinline__ void async_copy_b128(uint32_t lds_addr, const void* gptr) {
  asm volatile("global_load_async_to_lds_b128 %0, %1, off"
               :: "v"(lds_addr), "v"(gptr) : "memory");
}
__device__ __forceinline__ void async_copy_b128_off(uint32_t lds_addr, const void* gptr,
                                                    int boff) {
  asm volatile("global_load_async_to_lds_b128 %0, %1, off offset:%2"
               :: "v"(lds_addr), "v"(gptr), "i"(boff) : "memory");
}
__device__ __forceinline__ void wait_asynccnt0() {
  asm volatile("s_wait_asynccnt 0" ::: "memory");
}

// ---------- kernel 0: weights f32 -> f16 ----------
__global__ __launch_bounds__(256)
void wa_convert(const float* __restrict__ qkv_w, const float* __restrict__ proj_w,
                _Float16* __restrict__ wq, _Float16* __restrict__ wp) {
  const int i = blockIdx.x * blockDim.x + threadIdx.x;
  if (i < 1152 * 384) wq[i] = (_Float16)qkv_w[i];
  if (i < 384 * 384)  wp[i] = (_Float16)proj_w[i];
}

// ---------- kernel 1: QKV GEMM [ROWS,384] x [384,1152] -> q/k/v f16 ----------
// 128x128 block tile; grid (9 col tiles on x, 3136 row tiles on y); 256 thr = 8 waves.
// Wave w: M-tiles {2*(w&3), +1}, N-tiles 4*(w>>2)+{0..3} -> 8 accumulators.
__global__ __launch_bounds__(256)
void wa_qkv(const float* __restrict__ x, const _Float16* __restrict__ wq,
            const float* __restrict__ qkv_b,
            _Float16* __restrict__ qb, _Float16* __restrict__ kb,
            _Float16* __restrict__ vb) {
  __shared__ __align__(16) _Float16 at[2][128][AP];
  __shared__ __align__(16) _Float16 bt[2][128][AP];
  const int tid = threadIdx.x, lane = tid & 31, wave = tid >> 5;
  const int row0 = blockIdx.y * 128, col0 = blockIdx.x * 128;
  const int mt0 = (wave & 3) * 2, nt0 = (wave >> 2) * 4;

  // A staging: 128 rows x 32 f32 -> 16 floats per thread
  const int sr = tid >> 1, sc0 = (tid & 1) * 16;
  const float* xp = x + (size_t)(row0 + sr) * WA_C + sc0;

  // B tile: 128 rows x 32 halfs = 512 x 16B chunks; 2 chunks per thread
  const int bc0 = tid, bc1 = tid + 256;
  const int br0 = bc0 >> 2, bo0 = (bc0 & 3) * 8;
  const int br1 = bc1 >> 2, bo1 = (bc1 & 3) * 8;
  const _Float16* bsrc0 = wq + (size_t)(col0 + br0) * WA_C + bo0;
  const _Float16* bsrc1 = wq + (size_t)(col0 + br1) * WA_C + bo1;
  const uint32_t bl0[2] = { lds_of(&bt[0][br0][bo0]), lds_of(&bt[1][br0][bo0]) };
  const uint32_t bl1[2] = { lds_of(&bt[0][br1][bo1]), lds_of(&bt[1][br1][bo1]) };

  v8f acc[2][4] = {};
  float stage[16];
#pragma unroll
  for (int u = 0; u < 16; ++u) stage[u] = xp[u];
  async_copy_b128(bl0[0], bsrc0);
  async_copy_b128(bl1[0], bsrc1);

#pragma unroll
  for (int ks = 0; ks < 12; ++ks) {
    const int k0 = ks * 32;
    const int buf = ks & 1;
#pragma unroll
    for (int u = 0; u < 16; ++u) at[buf][sr][sc0 + u] = (_Float16)stage[u];
    wait_asynccnt0();
    __syncthreads();
    if (ks < 11) {
      async_copy_b128_off(bl0[buf ^ 1], bsrc0, (k0 + 32) * 2);
      async_copy_b128_off(bl1[buf ^ 1], bsrc1, (k0 + 32) * 2);
#pragma unroll
      for (int u = 0; u < 16; ++u) stage[u] = xp[k0 + 32 + u];  // prefetch next A
    }
    v16h a0 = frag_a(&at[buf][(mt0 + 0) * 16 + (lane & 15)][0], lane);
    v16h a1 = frag_a(&at[buf][(mt0 + 1) * 16 + (lane & 15)][0], lane);
    v16h bf[4];
#pragma unroll
    for (int p = 0; p < 4; ++p)
      bf[p] = frag_b(&bt[buf][(nt0 + p) * 16 + (lane & 15)][0], lane);
#pragma unroll
    for (int p = 0; p < 4; ++p) {
      acc[0][p] = wmma_f16(a0, bf[p], acc[0][p]);
      acc[1][p] = wmma_f16(a1, bf[p], acc[1][p]);
    }
  }

#pragma unroll
  for (int mi = 0; mi < 2; ++mi) {
#pragma unroll
    for (int p = 0; p < 4; ++p) {
      const int n = lane & 15;
      const int j = col0 + (nt0 + p) * 16 + n;   // one (s,head) per 16-wide tile
      const int s = j / WA_C;
      const int r = j - s * WA_C;
      const int head = r >> 5, dd = r & 31;
      _Float16* dst = (s == 0) ? qb : (s == 1) ? kb : vb;
      const float bias = qkv_b[j];
#pragma unroll
      for (int i = 0; i < 8; ++i) {
        const int m = (mt0 + mi) * 16 + i + ((lane & 16) ? 8 : 0);
        const int row = row0 + m;
        const int bw = row / WA_N;
        const int nn = row - bw * WA_N;
        dst[((size_t)(bw * WA_H + head) * WA_N + nn) * WA_D + dd] =
            (_Float16)(acc[mi][p][i] + bias);
      }
    }
  }
}

// ---------- kernel 2: attention per (window, head) ----------
__global__ __launch_bounds__(128)
void wa_attn(const _Float16* __restrict__ qb, const _Float16* __restrict__ kb,
             const _Float16* __restrict__ vb, const float* __restrict__ mask,
             const float* __restrict__ bias_table, const int* __restrict__ rel_idx,
             _Float16* __restrict__ ob) {
  __shared__ __align__(16) _Float16 qs[64][AP];
  __shared__ __align__(16) _Float16 ks[64][AP];
  __shared__ __align__(16) _Float16 vts[32][TP];  // v transposed
  __shared__ float                  sc[64][65];   // logits (scalar access)
  __shared__ __align__(16) _Float16 ps[64][TP];   // probabilities f16
  const int bh = blockIdx.x;
  const int bw = bh / WA_H, head = bh - bw * WA_H;
  const int tid = threadIdx.x, lane = tid & 31, mt = tid >> 5;

  const _Float16* qsrc = qb + (size_t)bh * WA_N * WA_D;
  const _Float16* ksrc = kb + (size_t)bh * WA_N * WA_D;
  const _Float16* vsrc = vb + (size_t)bh * WA_N * WA_D;

  // q/k: 49 rows x 4 chunks of 16B, async DMA into padded LDS rows
  for (int c = tid; c < 196; c += 128) {
    const int r = c >> 2, o = (c & 3) * 8;
    async_copy_b128(lds_of(&qs[r][o]), qsrc + c * 8);
    async_copy_b128(lds_of(&ks[r][o]), ksrc + c * 8);
  }
  // zero-pad rows 49..63
  for (int i = tid; i < 480; i += 128) {
    const int r = 49 + (i >> 5), c = i & 31;
    qs[r][c] = (_Float16)0.f;
    ks[r][c] = (_Float16)0.f;
  }
  // v transposed (VGPR path)
  for (int i = tid; i < 2048; i += 128) {
    const int r = i >> 5, c = i & 31;
    vts[c][r] = (r < WA_N) ? vsrc[r * 32 + c] : (_Float16)0.f;
  }
  wait_asynccnt0();
  __syncthreads();

  // S = q @ k^T
  {
    v16h a = frag_a(&qs[mt * 16 + (lane & 15)][0], lane);
    const int col_n = lane & 15;
    const float* mrow = mask + (size_t)(bw % WA_NW) * WA_N * WA_N;
#pragma unroll
    for (int nt = 0; nt < 4; ++nt) {
      v16h b = frag_b(&ks[nt * 16 + col_n][0], lane);
      v8f c = {};
      c = wmma_f16(a, b, c);
      const int col = nt * 16 + col_n;
#pragma unroll
      for (int i = 0; i < 8; ++i) {
        const int m = mt * 16 + i + ((lane & 16) ? 8 : 0);
        float logit = -1e30f;
        if (m < WA_N && col < WA_N) {
          logit = c[i] * WA_SCALE
                + bias_table[rel_idx[m * WA_N + col] * WA_H + head]
                + mrow[m * WA_N + col];
        }
        sc[m][col] = logit;
      }
    }
  }
  __syncthreads();

  // row softmax
  if (tid < 64) {
    if (tid < WA_N) {
      float mx = -1e30f;
      for (int c = 0; c < WA_N; ++c) mx = fmaxf(mx, sc[tid][c]);
      float sum = 0.f;
      for (int c = 0; c < 64; ++c) {
        const float e = __expf(sc[tid][c] - mx);
        sc[tid][c] = e;
        sum += e;
      }
      const float inv = 1.f / sum;
      for (int c = 0; c < 64; ++c) ps[tid][c] = (_Float16)(sc[tid][c] * inv);
    } else {
      for (int c = 0; c < 64; ++c) ps[tid][c] = (_Float16)0.f;
    }
  }
  __syncthreads();

  // O = P @ V
  {
    const int dn = lane & 15;
#pragma unroll
    for (int nt = 0; nt < 2; ++nt) {
      v8f c = {};
#pragma unroll
      for (int kk = 0; kk < 2; ++kk) {
        v16h a = frag_a(&ps[mt * 16 + (lane & 15)][kk * 32], lane);
        v16h b = frag_b(&vts[nt * 16 + dn][kk * 32], lane);
        c = wmma_f16(a, b, c);
      }
#pragma unroll
      for (int i = 0; i < 8; ++i) {
        const int m = mt * 16 + i + ((lane & 16) ? 8 : 0);
        if (m < WA_N) {
          ob[(size_t)(bw * WA_N + m) * WA_C + head * WA_D + nt * 16 + dn] =
              (_Float16)c[i];
        }
      }
    }
  }
}

// ---------- kernel 3: proj GEMM [ROWS,384] x [384,384] + bias -> f32 out ----------
// 128x128 block tile; A and B tiles async-DMA'd, double-buffered, K fully unrolled.
__global__ __launch_bounds__(256)
void wa_proj(const _Float16* __restrict__ ob, const _Float16* __restrict__ wp,
             const float* __restrict__ proj_b, float* __restrict__ out) {
  __shared__ __align__(16) _Float16 at[2][128][AP];
  __shared__ __align__(16) _Float16 bt[2][128][AP];
  const int tid = threadIdx.x, lane = tid & 31, wave = tid >> 5;
  const int row0 = blockIdx.y * 128, col0 = blockIdx.x * 128;
  const int mt0 = (wave & 3) * 2, nt0 = (wave >> 2) * 4;

  // A tile: 128 rows x 4 chunks -> 2 chunks per thread
  const int ac0 = tid, ac1 = tid + 256;
  const int ar0 = ac0 >> 2, ao0 = (ac0 & 3) * 8;
  const int ar1 = ac1 >> 2, ao1 = (ac1 & 3) * 8;
  const _Float16* asrc0 = ob + (size_t)(row0 + ar0) * WA_C + ao0;
  const _Float16* asrc1 = ob + (size_t)(row0 + ar1) * WA_C + ao1;
  const uint32_t al0[2] = { lds_of(&at[0][ar0][ao0]), lds_of(&at[1][ar0][ao0]) };
  const uint32_t al1[2] = { lds_of(&at[0][ar1][ao1]), lds_of(&at[1][ar1][ao1]) };
  // B tile: 128 rows x 4 chunks -> 2 chunks per thread
  const _Float16* bsrc0 = wp + (size_t)(col0 + ar0) * WA_C + ao0;
  const _Float16* bsrc1 = wp + (size_t)(col0 + ar1) * WA_C + ao1;
  const uint32_t bl0[2] = { lds_of(&bt[0][ar0][ao0]), lds_of(&bt[1][ar0][ao0]) };
  const uint32_t bl1[2] = { lds_of(&bt[0][ar1][ao1]), lds_of(&bt[1][ar1][ao1]) };

  v8f acc[2][4] = {};
  async_copy_b128(al0[0], asrc0);
  async_copy_b128(al1[0], asrc1);
  async_copy_b128(bl0[0], bsrc0);
  async_copy_b128(bl1[0], bsrc1);

#pragma unroll
  for (int ks = 0; ks < 12; ++ks) {
    const int k0 = ks * 32;
    const int buf = ks & 1;
    wait_asynccnt0();
    __syncthreads();
    if (ks < 11) {
      async_copy_b128_off(al0[buf ^ 1], asrc0, (k0 + 32) * 2);
      async_copy_b128_off(al1[buf ^ 1], asrc1, (k0 + 32) * 2);
      async_copy_b128_off(bl0[buf ^ 1], bsrc0, (k0 + 32) * 2);
      async_copy_b128_off(bl1[buf ^ 1], bsrc1, (k0 + 32) * 2);
    }
    v16h a0 = frag_a(&at[buf][(mt0 + 0) * 16 + (lane & 15)][0], lane);
    v16h a1 = frag_a(&at[buf][(mt0 + 1) * 16 + (lane & 15)][0], lane);
    v16h bf[4];
#pragma unroll
    for (int p = 0; p < 4; ++p)
      bf[p] = frag_b(&bt[buf][(nt0 + p) * 16 + (lane & 15)][0], lane);
#pragma unroll
    for (int p = 0; p < 4; ++p) {
      acc[0][p] = wmma_f16(a0, bf[p], acc[0][p]);
      acc[1][p] = wmma_f16(a1, bf[p], acc[1][p]);
    }
  }

#pragma unroll
  for (int mi = 0; mi < 2; ++mi) {
#pragma unroll
    for (int p = 0; p < 4; ++p) {
      const int n = lane & 15;
      const int j = col0 + (nt0 + p) * 16 + n;
      const float bias = proj_b[j];
#pragma unroll
      for (int i = 0; i < 8; ++i) {
        const int m = (mt0 + mi) * 16 + i + ((lane & 16) ? 8 : 0);
        out[(size_t)(row0 + m) * WA_C + j] = acc[mi][p][i] + bias;
      }
    }
  }
}

// ---------- host ----------
extern "C" void kernel_launch(void* const* d_in, const int* in_sizes, int n_in,
                              void* d_out, int out_size, void* d_ws, size_t ws_size,
                              hipStream_t stream) {
  const float* x          = (const float*)d_in[0];
  const float* mask       = (const float*)d_in[1];
  const float* qkv_w      = (const float*)d_in[2];
  const float* qkv_b      = (const float*)d_in[3];
  const float* proj_w     = (const float*)d_in[4];
  const float* proj_b     = (const float*)d_in[5];
  const float* bias_table = (const float*)d_in[6];
  const int*   rel_idx    = (const int*)d_in[7];
  float* out = (float*)d_out;

  char* ws = (char*)d_ws;
  size_t off = 0;
  auto walloc = [&](size_t bytes) -> char* {
    char* p = ws + off;
    off = (off + bytes + 255) & ~(size_t)255;
    return p;
  };
  _Float16* wq = (_Float16*)walloc(sizeof(_Float16) * 1152 * 384);
  _Float16* wp = (_Float16*)walloc(sizeof(_Float16) * 384 * 384);
  const size_t qkvElems = (size_t)WA_B * WA_H * WA_N * WA_D;
  _Float16* qb = (_Float16*)walloc(sizeof(_Float16) * qkvElems);
  _Float16* kb = (_Float16*)walloc(sizeof(_Float16) * qkvElems);
  _Float16* vb = (_Float16*)walloc(sizeof(_Float16) * qkvElems);
  _Float16* ob = (_Float16*)walloc(sizeof(_Float16) * (size_t)WA_ROWS * WA_C);

  wa_convert<<<dim3((1152 * 384 + 255) / 256), 256, 0, stream>>>(qkv_w, proj_w, wq, wp);
  wa_qkv<<<dim3((3 * WA_C) / 128, WA_ROWS / 128), 256, 0, stream>>>(x, wq, qkv_b, qb, kb, vb);
  wa_attn<<<dim3(WA_B * WA_H), 128, 0, stream>>>(qb, kb, vb, mask, bias_table, rel_idx, ob);
  wa_proj<<<dim3(WA_C / 128, WA_ROWS / 128), 256, 0, stream>>>(ob, wp, proj_b, out);
}
